// MaskedAttentionBlock_60498909331477
// MI455X (gfx1250) — matrix operand order repaired and checked by reading, not compiled, not run on hardware
//
#include <hip/hip_runtime.h>
#include <hip/hip_bf16.h>

// ---------------------------------------------------------------------------
// MaskedAttentionBlock for MI455X (gfx1250, wave32, WMMA + TDM)
//   B=8, S=2048, E=1024, fp32 in/out, bf16 WMMA internally (fp32 accum).
// GEMM data path: Tensor Data Mover stages 128x32 bf16 tiles (A and B) into
// padded LDS (double buffered, TENSORcnt-tracked), 8 waves consume fragments
// with ds_load_b128 and issue v_wmma_f32_16x16x32_bf16.
// ---------------------------------------------------------------------------

typedef __attribute__((ext_vector_type(16))) __bf16 v16bf;
typedef __attribute__((ext_vector_type(8)))  __bf16 v8bf;
typedef __attribute__((ext_vector_type(8)))  float  v8f;
typedef __attribute__((ext_vector_type(4)))  unsigned int v4u;
typedef __attribute__((ext_vector_type(8)))  int v8i_;
typedef __attribute__((ext_vector_type(4)))  int v4i_;

#define B_DIM 8
#define S_DIM 2048
#define E_DIM 1024

// LDS tile: 128 rows x 32 bf16, padded to 40 elements (80 B) per row so that
// 16 lanes reading 16B chunks at rows 0..15 hit distinct bank groups
// (20*r mod 64 is distinct for r=0..15).
#define TILE_ROWS 128
#define TILE_K    32
#define ROW_PAD   40

__device__ __forceinline__ unsigned short f2bf_bits(float f) {
    unsigned int u = __float_as_uint(f);
    unsigned int r = (u + 0x7FFFu + ((u >> 16) & 1u)) >> 16;   // RNE
    return (unsigned short)r;
}

__device__ __forceinline__ unsigned lds_byte_off(const void* p) {
    // For LDS apertures the low 32 bits of the flat address are the LDS offset.
    return (unsigned)(unsigned long long)p;
}

// Issue one TDM load of a 128 x 32 bf16 tile (row stride `ld` elements) from
// `gptr` into LDS at byte offset `ldsoff`, padding 4 DWORDs per 16 DWORDs
// (per-row) to produce the 80-byte padded row layout.
__device__ __forceinline__ void tdm_load_tile(const unsigned short* gptr,
                                              unsigned ldsoff, int ld) {
    unsigned long long ga = (unsigned long long)gptr;
    v4u g0;
    g0[0] = 1u;                                             // count=1, user mode
    g0[1] = ldsoff;                                         // lds_addr (bytes)
    g0[2] = (unsigned)ga;                                   // global_addr[31:0]
    g0[3] = (unsigned)((ga >> 32) & 0x01FFFFFFull) | (2u << 30); // addr[56:32] | type=2

    v8i_ g1;
    // wg_mask=0, data_size=2B, pad_enable, pad_interval=16 DW, pad_amount=4 DW
    g1[0] = (int)0x06D10000u;
    g1[1] = (int)(((unsigned)ld & 0xFFFFu) << 16);          // tensor_dim0[15:0]
    g1[2] = (int)((((unsigned)ld >> 16) & 0xFFFFu)          // tensor_dim0[31:16]
                  | ((unsigned)TILE_ROWS << 16));           // tensor_dim1[15:0]
    g1[3] = (int)((unsigned)TILE_K << 16);                  // tensor_dim1 hi=0 | tile_dim0
    g1[4] = TILE_ROWS;                                      // tile_dim1 | tile_dim2=0
    g1[5] = ld;                                             // tensor_dim0_stride[31:0]
    g1[6] = 0;                                              // stride hi | dim1_stride lo
    g1[7] = 0;

    v4i_ zz = {0, 0, 0, 0};
#if __clang_major__ >= 23
    v8i_ z8 = {0, 0, 0, 0, 0, 0, 0, 0};
    __builtin_amdgcn_tensor_load_to_lds(g0, g1, zz, zz, z8, 0);
#else
    __builtin_amdgcn_tensor_load_to_lds(g0, g1, zz, zz, 0);
#endif
}

// ------------------------------ cast kernel --------------------------------
__global__ __launch_bounds__(256)
void f32_to_bf16_kernel(const float* __restrict__ in, unsigned short* __restrict__ out,
                        long long n) {
    long long i = (long long)blockIdx.x * blockDim.x + threadIdx.x;
    long long stride = (long long)gridDim.x * blockDim.x;
    for (; i < n; i += stride) out[i] = f2bf_bits(in[i]);
}

// ------------------------------ GEMM (NT) ----------------------------------
// D[m,n] = sum_k A[m,k]*B[n,k]   (both K-contiguous, bf16), fp32 accumulate.
// Block: 256 thr (8 waves). Block tile 128x128. Wave tile 64x32 (4x2 WMMA).
// Wave 0 drives the TDM double-buffer pipeline; all waves compute from LDS.
enum { EPI_F32 = 0, EPI_QK = 1, EPI_VT = 2 };

template<int EPI, bool CAUSAL>
__global__ __launch_bounds__(256)
void gemm_nt_bf16(const unsigned short* __restrict__ A, long long strideAb, int lda,
                  const unsigned short* __restrict__ B, long long strideBb, int ldb,
                  void* __restrict__ C, long long strideCb, int ldc,
                  const float* __restrict__ bias, float scale, int K) {
    const int b  = blockIdx.z;
    const int bm = blockIdx.y * 128;
    const int bn = blockIdx.x * 128;
    if (CAUSAL && bn > bm + 127) return;   // tile entirely above causal diagonal

    __shared__ __align__(16) unsigned short smA[2][TILE_ROWS * ROW_PAD];
    __shared__ __align__(16) unsigned short smB[2][TILE_ROWS * ROW_PAD];

    const unsigned short* Ablk = A + (long long)b * strideAb + (long long)bm * lda;
    const unsigned short* Bblk = B + (long long)b * strideBb + (long long)bn * ldb;

    const int tid  = threadIdx.x;
    const int wave = tid >> 5;
    const int lane = tid & 31;
    const int h    = lane >> 4;    // half of the wave
    const int r    = lane & 15;    // row (A) / col (B) within 16x16 tile
    const int wm   = (wave >> 2) * 64;   // 0 or 64
    const int wn   = (wave & 3) * 32;    // 0,32,64,96
    const bool issuer = (wave == 0);     // wave-uniform: one TDM issuer per block

    const int nk = K / TILE_K;

    if (issuer) {
        tdm_load_tile(Ablk, lds_byte_off(&smA[0][0]), lda);
        tdm_load_tile(Bblk, lds_byte_off(&smB[0][0]), ldb);
    }

    v8f acc[4][2] = {};

    for (int kt = 0; kt < nk; ++kt) {
        const int buf = kt & 1;
        if (issuer) {
            if (kt + 1 < nk) {   // prefetch next pair into the other buffer
                tdm_load_tile(Ablk + (long long)(kt + 1) * TILE_K,
                              lds_byte_off(&smA[buf ^ 1][0]), lda);
                tdm_load_tile(Bblk + (long long)(kt + 1) * TILE_K,
                              lds_byte_off(&smB[buf ^ 1][0]), ldb);
                __builtin_amdgcn_s_wait_tensorcnt(2);  // current pair complete
            } else {
                __builtin_amdgcn_s_wait_tensorcnt(0);
            }
        }
        __syncthreads();   // publish LDS tile to all waves

        const unsigned short* As = &smA[buf][0];
        const unsigned short* Bs = &smB[buf][0];

        // B fragments: col n = r, 16 contiguous K at 16*h (padded row stride)
        v16bf bfrag[2];
#pragma unroll
        for (int j = 0; j < 2; ++j) {
            const unsigned short* bp = Bs + (wn + 16 * j + r) * ROW_PAD + 16 * h;
            v8bf lo = *(const v8bf*)bp;
            v8bf hi = *(const v8bf*)(bp + 8);
            bfrag[j] = __builtin_shufflevector(lo, hi,
                0,1,2,3,4,5,6,7,8,9,10,11,12,13,14,15);
        }
        // A fragments: row m = r, 16 contiguous K at 16*h
        v16bf afrag[4];
#pragma unroll
        for (int i = 0; i < 4; ++i) {
            const unsigned short* ap = As + (wm + 16 * i + r) * ROW_PAD + 16 * h;
            v8bf lo = *(const v8bf*)ap;
            v8bf hi = *(const v8bf*)(ap + 8);
            afrag[i] = __builtin_shufflevector(lo, hi,
                0,1,2,3,4,5,6,7,8,9,10,11,12,13,14,15);
        }
#pragma unroll
        for (int i = 0; i < 4; ++i)
#pragma unroll
            for (int j = 0; j < 2; ++j)
                acc[i][j] = __builtin_amdgcn_wmma_f32_16x16x32_bf16(
                    false, afrag[i], false, bfrag[j],
                    (short)0, acc[i][j], false, false);

        __syncthreads();   // tile fully consumed before issuer overwrites it
    }

    // Epilogue. C/D layout: VGPR v, lane -> (m = v + 8*h, n = r) within tile.
    const long long cbase = (long long)b * strideCb;
#pragma unroll
    for (int i = 0; i < 4; ++i) {
#pragma unroll
        for (int j = 0; j < 2; ++j) {
            const int m0 = bm + wm + 16 * i;
            const int n0 = bn + wn + 16 * j;
#pragma unroll
            for (int v = 0; v < 8; ++v) {
                const int m = m0 + v + 8 * h;
                const int n = n0 + r;
                float val = acc[i][j][v];
                if constexpr (EPI == EPI_F32) {
                    ((float*)C)[cbase + (long long)m * ldc + n] = val;
                } else if constexpr (EPI == EPI_QK) {
                    val = (val + bias[n]) * scale;
                    ((unsigned short*)C)[cbase + (long long)m * ldc + n] = f2bf_bits(val);
                } else { // EPI_VT: transposed store Vt[n][m], ldc = S
                    val = val + bias[n];
                    ((unsigned short*)C)[cbase + (long long)n * ldc + m] = f2bf_bits(val);
                }
            }
        }
    }
}

// --------------------------- causal softmax --------------------------------
__device__ __forceinline__ float block_reduce_max(float v, float* red, int tid) {
#pragma unroll
    for (int off = 16; off > 0; off >>= 1) v = fmaxf(v, __shfl_xor(v, off, 32));
    if ((tid & 31) == 0) red[tid >> 5] = v;
    __syncthreads();
    float m = red[0];
#pragma unroll
    for (int w = 1; w < 8; ++w) m = fmaxf(m, red[w]);
    __syncthreads();
    return m;
}
__device__ __forceinline__ float block_reduce_sum(float v, float* red, int tid) {
#pragma unroll
    for (int off = 16; off > 0; off >>= 1) v += __shfl_xor(v, off, 32);
    if ((tid & 31) == 0) red[tid >> 5] = v;
    __syncthreads();
    float s = 0.f;
#pragma unroll
    for (int w = 0; w < 8; ++w) s += red[w];
    __syncthreads();
    return s;
}

__global__ __launch_bounds__(256)
void softmax_causal_kernel(const float* __restrict__ scores,
                           unsigned short* __restrict__ P) {
    const int row = blockIdx.x;            // b*S + q
    const int q   = row & (S_DIM - 1);
    const float* srow = scores + (long long)row * S_DIM;
    unsigned short* prow = P + (long long)row * S_DIM;
    const int tid = threadIdx.x;
    __shared__ float red[8];

    float vals[S_DIM / 256];
    float lmax = -__builtin_inff();
#pragma unroll
    for (int it = 0; it < S_DIM / 256; ++it) {
        const int k = it * 256 + tid;
        float v = (k <= q) ? srow[k] : -__builtin_inff();
        vals[it] = v;
        lmax = fmaxf(lmax, v);
    }
    const float bmax = block_reduce_max(lmax, red, tid);

    float lsum = 0.f;
#pragma unroll
    for (int it = 0; it < S_DIM / 256; ++it) {
        float e = __expf(vals[it] - bmax);   // exp(-inf)=0 above diagonal
        vals[it] = e;
        lsum += e;
    }
    const float inv = 1.0f / block_reduce_sum(lsum, red, tid);
#pragma unroll
    for (int it = 0; it < S_DIM / 256; ++it) {
        const int k = it * 256 + tid;
        prow[k] = f2bf_bits(vals[it] * inv);
    }
}

// ------------------------ residual + layernorm -----------------------------
__global__ __launch_bounds__(256)
void add_layernorm_kernel(const float* __restrict__ O, const float* __restrict__ X,
                          const float* __restrict__ gamma, const float* __restrict__ beta,
                          float* __restrict__ out) {
    const int row = blockIdx.x;            // b*S + s
    const float* orow = O + (long long)row * E_DIM;
    const float* xrow = X + (long long)row * E_DIM;
    float* yrow = out + (long long)row * E_DIM;
    const int tid = threadIdx.x;
    __shared__ float red[8];

    float x[E_DIM / 256];
    float s = 0.f, s2 = 0.f;
#pragma unroll
    for (int it = 0; it < E_DIM / 256; ++it) {
        const int e = it * 256 + tid;
        float v = orow[e] + xrow[e];
        x[it] = v;
        s += v;
        s2 += v * v;
    }
    s  = block_reduce_sum(s,  red, tid);
    s2 = block_reduce_sum(s2, red, tid);
    const float mu   = s * (1.0f / E_DIM);
    const float var  = s2 * (1.0f / E_DIM) - mu * mu;
    const float rstd = rsqrtf(var + 1e-5f);
#pragma unroll
    for (int it = 0; it < E_DIM / 256; ++it) {
        const int e = it * 256 + tid;
        yrow[e] = (x[it] - mu) * rstd * gamma[e] + beta[e];
    }
}

// ------------------------------ launcher -----------------------------------
extern "C" void kernel_launch(void* const* d_in, const int* in_sizes, int n_in,
                              void* d_out, int out_size, void* d_ws, size_t ws_size,
                              hipStream_t stream) {
    const float* emb   = (const float*)d_in[0];
    const float* Wq    = (const float*)d_in[1];
    const float* bq    = (const float*)d_in[2];
    const float* Wk    = (const float*)d_in[3];
    const float* bk    = (const float*)d_in[4];
    const float* Wv    = (const float*)d_in[5];
    const float* bv    = (const float*)d_in[6];
    const float* gamma = (const float*)d_in[7];
    const float* beta  = (const float*)d_in[8];

    const long long nX  = (long long)B_DIM * S_DIM * E_DIM;   // 16,777,216
    const long long nW  = (long long)E_DIM * E_DIM;           //  1,048,576
    const long long nSS = (long long)B_DIM * S_DIM * S_DIM;   // 33,554,432

    // Workspace layout (bytes)
    char* w = (char*)d_ws;
    unsigned short* Xb  = (unsigned short*)w;                 w += nX  * 2;
    unsigned short* Wqb = (unsigned short*)w;                 w += nW  * 2;
    unsigned short* Wkb = (unsigned short*)w;                 w += nW  * 2;
    unsigned short* Wvb = (unsigned short*)w;                 w += nW  * 2;
    unsigned short* Qb  = (unsigned short*)w;                 w += nX  * 2;
    unsigned short* Kb  = (unsigned short*)w;                 w += nX  * 2;
    unsigned short* Vt  = (unsigned short*)w;                 w += nX  * 2;  // [b][e][s]
    float*          Sc  = (float*)w;                          w += nSS * 4;
    unsigned short* Pb  = (unsigned short*)w;                 w += nSS * 2;
    float*          Of  = (float*)w;                          w += nX  * 4;

    // 1) casts
    f32_to_bf16_kernel<<<8192, 256, 0, stream>>>(emb, Xb, nX);
    f32_to_bf16_kernel<<<512,  256, 0, stream>>>(Wq,  Wqb, nW);
    f32_to_bf16_kernel<<<512,  256, 0, stream>>>(Wk,  Wkb, nW);
    f32_to_bf16_kernel<<<512,  256, 0, stream>>>(Wv,  Wvb, nW);

    const long long sSE = (long long)S_DIM * E_DIM;
    const long long sSS = (long long)S_DIM * S_DIM;
    const float qscale = 0.03125f;  // 1/sqrt(1024)

    // 2) QKV projections, batched over z = B
    dim3 gQKV(E_DIM / 128, S_DIM / 128, B_DIM);
    gemm_nt_bf16<EPI_QK, false><<<gQKV, 256, 0, stream>>>(
        Xb, sSE, E_DIM, Wqb, 0, E_DIM, Qb, sSE, E_DIM, bq, qscale, E_DIM);
    gemm_nt_bf16<EPI_QK, false><<<gQKV, 256, 0, stream>>>(
        Xb, sSE, E_DIM, Wkb, 0, E_DIM, Kb, sSE, E_DIM, bk, 1.0f, E_DIM);
    gemm_nt_bf16<EPI_VT, false><<<gQKV, 256, 0, stream>>>(
        Xb, sSE, E_DIM, Wvb, 0, E_DIM, Vt, sSE, S_DIM, bv, 1.0f, E_DIM);

    // 3) scores = Q K^T (Q pre-scaled), fp32, skip fully-masked tiles
    dim3 gSc(S_DIM / 128, S_DIM / 128, B_DIM);
    gemm_nt_bf16<EPI_F32, true><<<gSc, 256, 0, stream>>>(
        Qb, sSE, E_DIM, Kb, sSE, E_DIM, Sc, sSS, S_DIM, nullptr, 1.0f, E_DIM);

    // 4) causal softmax -> P bf16
    softmax_causal_kernel<<<B_DIM * S_DIM, 256, 0, stream>>>(Sc, Pb);

    // 5) O = P Vt^T, fp32
    dim3 gO(E_DIM / 128, S_DIM / 128, B_DIM);
    gemm_nt_bf16<EPI_F32, false><<<gO, 256, 0, stream>>>(
        Pb, sSS, S_DIM, Vt, sSE, S_DIM, Of, sSE, E_DIM, nullptr, 1.0f, S_DIM);

    // 6) residual + layernorm
    add_layernorm_kernel<<<B_DIM * S_DIM, 256, 0, stream>>>(
        Of, emb, gamma, beta, (float*)d_out);
}
